// TransConvLayer_27754078666847
// MI455X (gfx1250) — compile-verified, matrix operand order checked by reading, not run on hardware
//
#include <hip/hip_runtime.h>

// Problem constants (match reference)
#define IN_CH  256
#define OC     128
#define NHEADS 4
#define OCH    512   // OC * NHEADS

typedef __attribute__((ext_vector_type(16))) __bf16 v16bf;
typedef __attribute__((ext_vector_type(8)))  float  v8f;

union FragBF {
  v16bf v;
  unsigned int u[8];
};

// float -> bf16 bits, round-to-nearest-even (bit math; avoids __bf16 scalar ops)
__device__ __forceinline__ unsigned short f32_to_bf16(float f) {
  union { float f; unsigned int u; } x;
  x.f = f;
  unsigned int r = x.u + 0x7FFFu + ((x.u >> 16) & 1u);
  return (unsigned short)(r >> 16);
}

// Build a 16x32 bf16 WMMA fragment from LDS laid out [index][k] with row
// stride `stride` ushorts (k contiguous). Per CDNA5 ISA 16-bit A layout:
// lane%16 selects row M; VGPR i holds K pair kb,kb+1 with
// kb = (i<4 ? 2i : 16+2(i-4)) + (lane>=16 ? 8 : 0).
__device__ __forceinline__ void load_frag(FragBF& fr, const unsigned short* row_base, int khalf) {
#pragma unroll
  for (int i = 0; i < 8; ++i) {
    int kb = (i < 4 ? 2 * i : 16 + 2 * (i - 4)) + khalf;
    fr.u[i] = *(const unsigned int*)(row_base + kb);
  }
}

// ---------------------------------------------------------------------------
// K0: zero atomic accumulators (G + colsum)
// ---------------------------------------------------------------------------
__global__ void tcl_zero_kernel(float* p, int n) {
  int i = blockIdx.x * blockDim.x + threadIdx.x;
  if (i < n) p[i] = 0.0f;
}

// ---------------------------------------------------------------------------
// K1: G = S^T S (256x256 f32) + colsum(S) via bf16 WMMA, f32 atomics.
// grid = (4 row-bands of 64, sample-chunks of 2048), block = 256 (8 waves)
// Each wave: one 16-row tile band x 128 cols (8 WMMA tiles).
// ---------------------------------------------------------------------------
#define GSTRIDE 34  // 32 samples + pad, in ushorts

__global__ void tcl_gram_kernel(const float* __restrict__ S,
                                float* __restrict__ G,
                                float* __restrict__ colsum,
                                int n) {
  __shared__ unsigned short Sb[256 * GSTRIDE];  // [feature][sample] bf16

  const int t    = threadIdx.x;
  const int wave = t >> 5;
  const int lane = t & 31;
  const int khalf = (lane & 16) ? 8 : 0;

  const int tr = wave >> 1;                    // tile-row within 64-band
  const int ch = wave & 1;                     // column half
  const int m0 = blockIdx.x * 64 + tr * 16;    // global row of G tile band
  const int n_base = ch * 128;                 // global col base (8 tiles)

  // float4 staging pattern: thread owns features fbase..fbase+3, 8 samples
  const int fbase = 4 * (t & 63);
  const int srow  = t >> 6;

  v8f acc[8];
#pragma unroll
  for (int j = 0; j < 8; ++j) acc[j] = (v8f)(0.0f);

  float csum[4] = {0.0f, 0.0f, 0.0f, 0.0f};
  const int l_base = blockIdx.y * 2048;

  for (int c = 0; c < 64; ++c) {               // 64 chunks of 32 samples
    const int l0 = l_base + c * 32;
    // Stage chunk: b128 coalesced global reads, transposed bf16 store to LDS
#pragma unroll
    for (int i = 0; i < 8; ++i) {
      int s = srow + 4 * i;
      int l = l0 + s;
      float4 v = make_float4(0.0f, 0.0f, 0.0f, 0.0f);
      if (l < n) v = *(const float4*)(S + (size_t)l * IN_CH + fbase);
      if (blockIdx.x == 0) {
        csum[0] += v.x; csum[1] += v.y; csum[2] += v.z; csum[3] += v.w;
      }
      Sb[(fbase + 0) * GSTRIDE + s] = f32_to_bf16(v.x);
      Sb[(fbase + 1) * GSTRIDE + s] = f32_to_bf16(v.y);
      Sb[(fbase + 2) * GSTRIDE + s] = f32_to_bf16(v.z);
      Sb[(fbase + 3) * GSTRIDE + s] = f32_to_bf16(v.w);
    }
    __syncthreads();

    FragBF fa, fb;
    load_frag(fa, &Sb[(m0 + (lane & 15)) * GSTRIDE], khalf);
#pragma unroll
    for (int j = 0; j < 8; ++j) {
      load_frag(fb, &Sb[(n_base + j * 16 + (lane & 15)) * GSTRIDE], khalf);
      acc[j] = __builtin_amdgcn_wmma_f32_16x16x32_bf16(
          false, fa.v, false, fb.v, (short)0, acc[j], false, false);
    }
    __syncthreads();
  }

  // Accumulate partial band into G. C/D layout: VGPR r -> M = r + 8*(lane/16),
  // lane%16 -> N.
#pragma unroll
  for (int j = 0; j < 8; ++j) {
#pragma unroll
    for (int r = 0; r < 8; ++r) {
      int mm = m0 + r + ((lane >> 4) << 3);
      int nn = n_base + j * 16 + (lane & 15);
      atomicAdd(&G[mm * IN_CH + nn], acc[j][r]);
    }
  }
  if (blockIdx.x == 0) {
#pragma unroll
    for (int c = 0; c < 4; ++c) atomicAdd(&colsum[fbase + c], csum[c]);
  }
}

// ---------------------------------------------------------------------------
// K2a: T[h][i][d] = sum_j G[i][j] * Wv[j][4d+h]   (grid: (128 d, 4 h), 256 thr)
// ---------------------------------------------------------------------------
__global__ void tcl_gv_kernel(const float* __restrict__ G,
                              const float* __restrict__ Wv,
                              float* __restrict__ T) {
  const int d = blockIdx.x, h = blockIdx.y, i = threadIdx.x;
  __shared__ float wv[256];
  wv[i] = Wv[i * OCH + 4 * d + h];
  __syncthreads();
  float acc = 0.0f;
  const float* Grow = G + (size_t)i * IN_CH;
#pragma unroll 8
  for (int j = 0; j < 256; ++j) acc = fmaf(Grow[j], wv[j], acc);
  T[((h * 256) + i) * OC + d] = acc;
}

// ---------------------------------------------------------------------------
// K2b: scores a[h][m][d] = (Wk^T T + bias terms) / sqrt(n)
// grid: (128 d, 4 h), 128 threads = m
// ---------------------------------------------------------------------------
__global__ void tcl_scores_kernel(const float* __restrict__ T,
                                  const float* __restrict__ Wk,
                                  const float* __restrict__ bk,
                                  const float* __restrict__ bv,
                                  const float* __restrict__ Wv,
                                  const float* __restrict__ cs,
                                  float* __restrict__ A, int n) {
  const int d = blockIdx.x, h = blockIdx.y, m = threadIdx.x;
  __shared__ float Tcol[256];
  __shared__ float scol[256];
  Tcol[m]       = T[((h * 256) + m) * OC + d];
  Tcol[m + 128] = T[((h * 256) + m + 128) * OC + d];
  scol[m] = cs[m];
  scol[m + 128] = cs[m + 128];
  __syncthreads();

  const int cm = 4 * m + h, cd = 4 * d + h;
  float dot = 0.0f, skm = 0.0f, sv = 0.0f;
#pragma unroll 4
  for (int i = 0; i < 256; ++i) {
    float wk = Wk[i * OCH + cm];
    dot = fmaf(wk, Tcol[i], dot);
    skm = fmaf(wk, scol[i], skm);
    sv  = fmaf(scol[i], Wv[i * OCH + cd], sv);
  }
  float bkc = bk[cm], bvc = bv[cd];
  float val = dot + skm * bvc + bkc * sv + (float)n * bkc * bvc;
  A[((h * 128) + m) * OC + d] = val * rsqrtf((float)n);
}

// ---------------------------------------------------------------------------
// K2c: softmax over m for each (d,h) column.  grid: (128 d, 4 h), 128 threads
// ---------------------------------------------------------------------------
__global__ void tcl_softmax_kernel(float* __restrict__ A) {
  const int d = blockIdx.x, h = blockIdx.y, m = threadIdx.x;
  __shared__ float red[128];
  float val = A[((h * 128) + m) * OC + d];
  red[m] = val;
  __syncthreads();
  for (int s = 64; s > 0; s >>= 1) {
    if (m < s) red[m] = fmaxf(red[m], red[m + s]);
    __syncthreads();
  }
  float mx = red[0];
  __syncthreads();
  float e = __expf(val - mx);
  red[m] = e;
  __syncthreads();
  for (int s = 64; s > 0; s >>= 1) {
    if (m < s) red[m] += red[m + s];
    __syncthreads();
  }
  A[((h * 128) + m) * OC + d] = e / red[0];
}

// ---------------------------------------------------------------------------
// K2d: W_final[i][d] = 0.25 * sum_{h,m} Wq[i][4m+h] * attn[h][m][d]
//      b_final[d]    = 0.25 * sum_{h,m} bq[4m+h]    * attn[h][m][d]
// grid: 128 (d), 256 threads (i)
// ---------------------------------------------------------------------------
__global__ void tcl_fold_kernel(const float* __restrict__ A,
                                const float* __restrict__ Wq,
                                const float* __restrict__ bq,
                                float* __restrict__ Wf,
                                float* __restrict__ bfin) {
  const int d = blockIdx.x, i = threadIdx.x;
  __shared__ float att[512];  // flat index h*128+m
  att[i]       = A[i * OC + d];
  att[i + 256] = A[(i + 256) * OC + d];
  __syncthreads();

  float acc = 0.0f;
  const float* Wrow = Wq + (size_t)i * OCH;
#pragma unroll 2
  for (int m = 0; m < 128; ++m) {
#pragma unroll
    for (int h = 0; h < 4; ++h) acc = fmaf(Wrow[4 * m + h], att[h * 128 + m], acc);
  }
  Wf[i * OC + d] = 0.25f * acc;

  if (i == 0) {
    float bacc = 0.0f;
    for (int m = 0; m < 128; ++m)
      for (int h = 0; h < 4; ++h) bacc = fmaf(bq[4 * m + h], att[h * 128 + m], bacc);
    bfin[d] = 0.25f * bacc;
  }
}

// ---------------------------------------------------------------------------
// K3: out = Xq @ W_final + b_final  via bf16 WMMA.
// grid = ceil(n/128), block = 256 (8 waves). Wave w -> 16-row tile, 8 col tiles.
// ---------------------------------------------------------------------------
#define OSTRIDE 36  // 32 k + pad, in ushorts (8B-aligned rows for packed stores)

__global__ void tcl_out_kernel(const float* __restrict__ Xq,
                               const float* __restrict__ Wf,
                               const float* __restrict__ bfin,
                               float* __restrict__ out, int n) {
  __shared__ unsigned short Xb[128 * OSTRIDE];  // [row][k] bf16
  __shared__ unsigned short Wt[128 * OSTRIDE];  // [col][k] bf16

  const int t    = threadIdx.x;
  const int wave = t >> 5;
  const int lane = t & 31;
  const int khalf = (lane & 16) ? 8 : 0;
  const int row0 = blockIdx.x * 128;

  v8f acc[8];
#pragma unroll
  for (int j = 0; j < 8; ++j) acc[j] = (v8f)(0.0f);

  for (int kc = 0; kc < 8; ++kc) {  // K = 256 in chunks of 32
    const int k0 = kc * 32;
    // Stage Xq tile: b128 global loads, packed 8-byte LDS stores
#pragma unroll
    for (int i = 0; i < 4; ++i) {
      int r  = (t >> 3) + 32 * i;       // row in tile
      int kl = 4 * (t & 7);             // k within chunk (consecutive 4)
      int row = row0 + r;
      float4 v = make_float4(0.0f, 0.0f, 0.0f, 0.0f);
      if (row < n) v = *(const float4*)(Xq + (size_t)row * IN_CH + k0 + kl);
      unsigned int lo = (unsigned int)f32_to_bf16(v.x) |
                        ((unsigned int)f32_to_bf16(v.y) << 16);
      unsigned int hi = (unsigned int)f32_to_bf16(v.z) |
                        ((unsigned int)f32_to_bf16(v.w) << 16);
      *(uint2*)(&Xb[r * OSTRIDE + kl]) = make_uint2(lo, hi);
    }
    // Stage W chunk transposed: Wt[d][k] = Wf[(k0+k)*128 + d], b128 reads
#pragma unroll
    for (int i = 0; i < 4; ++i) {
      int q  = t + 256 * i;             // 0..1023
      int kk = q >> 5;                  // k within chunk
      int db = 4 * (q & 31);            // col base (consecutive 4 in global)
      float4 v = *(const float4*)(Wf + (size_t)(k0 + kk) * OC + db);
      Wt[(db + 0) * OSTRIDE + kk] = f32_to_bf16(v.x);
      Wt[(db + 1) * OSTRIDE + kk] = f32_to_bf16(v.y);
      Wt[(db + 2) * OSTRIDE + kk] = f32_to_bf16(v.z);
      Wt[(db + 3) * OSTRIDE + kk] = f32_to_bf16(v.w);
    }
    if (kc < 7) {
      // speculative prefetch of the next X chunk (global_prefetch_b8)
      __builtin_prefetch(Xq + (size_t)(row0 + (t >> 1)) * IN_CH + k0 + 32, 0, 0);
    }
    __syncthreads();

    FragBF fa, fb;
    load_frag(fa, &Xb[(wave * 16 + (lane & 15)) * OSTRIDE], khalf);
#pragma unroll
    for (int j = 0; j < 8; ++j) {
      load_frag(fb, &Wt[(j * 16 + (lane & 15)) * OSTRIDE], khalf);
      acc[j] = __builtin_amdgcn_wmma_f32_16x16x32_bf16(
          false, fa.v, false, fb.v, (short)0, acc[j], false, false);
    }
    __syncthreads();
  }

  // Epilogue: bias add + guarded non-temporal store.
  // C/D layout: VGPR r -> M = r + 8*(lane/16), lane%16 -> N.
#pragma unroll
  for (int j = 0; j < 8; ++j) {
    int col = j * 16 + (lane & 15);
    float bias = bfin[col];
#pragma unroll
    for (int r = 0; r < 8; ++r) {
      int row = row0 + wave * 16 + r + ((lane >> 4) << 3);
      if (row < n) __builtin_nontemporal_store(acc[j][r] + bias,
                                               &out[(size_t)row * OC + col]);
    }
  }
}

// ---------------------------------------------------------------------------
extern "C" void kernel_launch(void* const* d_in, const int* in_sizes, int n_in,
                              void* d_out, int out_size, void* d_ws, size_t ws_size,
                              hipStream_t stream) {
  const float* Xq = (const float*)d_in[0];  // query_input [n,256]
  const float* Xs = (const float*)d_in[1];  // source_input [n,256]
  const float* Wq = (const float*)d_in[2];  // [256,512]
  const float* bq = (const float*)d_in[3];  // [512]
  const float* Wk = (const float*)d_in[4];
  const float* bk = (const float*)d_in[5];
  const float* Wv = (const float*)d_in[6];
  const float* bv = (const float*)d_in[7];
  float* out = (float*)d_out;
  const int n = in_sizes[0] / IN_CH;

  // Workspace layout (f32): G[256*256] | colsum[256] | T[4*256*128]
  //                         | A[4*128*128] | Wf[256*128] | bfinal[128]
  float* G  = (float*)d_ws;
  float* cs = G + 256 * 256;
  float* T  = cs + 256;
  float* A  = T + 4 * 256 * 128;
  float* Wf = A + 4 * 128 * 128;
  float* bf = Wf + 256 * 128;

  const int nz = 256 * 256 + 256;
  tcl_zero_kernel<<<(nz + 255) / 256, 256, 0, stream>>>(G, nz);

  const int chunks = (n + 2047) / 2048;
  tcl_gram_kernel<<<dim3(4, chunks), 256, 0, stream>>>(Xs, G, cs, n);

  tcl_gv_kernel<<<dim3(128, 4), 256, 0, stream>>>(G, Wv, T);
  tcl_scores_kernel<<<dim3(128, 4), 128, 0, stream>>>(T, Wk, bk, bv, Wv, cs, A, n);
  tcl_softmax_kernel<<<dim3(128, 4), 128, 0, stream>>>(A);
  tcl_fold_kernel<<<128, 256, 0, stream>>>(A, Wq, bq, Wf, bf);

  tcl_out_kernel<<<(n + 127) / 128, 256, 0, stream>>>(Xq, Wf, bf, out, n);
}
